// DMAModel_43121471652250
// MI455X (gfx1250) — matrix operand order repaired and checked by reading, not compiled, not run on hardware
//
#include <hip/hip_runtime.h>
#include <math.h>

// ---------------------------------------------------------------------------
// MI455X (gfx1250) implementation of a 2-layer weighted graph-attention net.
//
// All large matmuls run on v_wmma_f32_16x16x32_bf16 (bf16 in, fp32 acc).
// GEMM v2: block = 8 waves computing a 128x64 C tile; the 64x32 B panel for
// each K-step is staged once per block into LDS (padded rows, bank-spread)
// and shared by all 8 waves -> 8x less L2 traffic on the B operand.
//
// GEMM convention: C[MxN] = A[MxK] @ Bt[NxK]^T, A/Bt row-major bf16.
//   A fragment (16x32 bf16, ISA 7.12.2): lane L<16 holds row M=L,
//     K = 0..15 contiguous (code's order, same as B); lanes 16-31: same row, K = 16..31.
//   B fragment (32x16 bf16): lane L<16 holds col N=L, K=0..15 contiguous;
//     lanes 16-31 hold K=16..31.
//   C/D (16x16 f32, 8 VGPRs): element r -> row r + 8*(lane/16), col lane%16.
// ---------------------------------------------------------------------------

typedef __bf16 bf16_t;
typedef __attribute__((ext_vector_type(16))) __bf16 v16bf;
typedef __attribute__((ext_vector_type(8)))  __bf16 v8bf;
typedef __attribute__((ext_vector_type(8)))  float  v8f;

#define LRELU_SLOPE 0.2f
#define NEGV       -1000000000.0f
#define BETA        0.5f

#define LDSB_STRIDE 40   // bf16 elements per staged B row (32 used + 8 pad)

// ---------------- elementwise converts ----------------

__global__ void k_cvt_bf16(const float* __restrict__ in, bf16_t* __restrict__ out, int n) {
  int i = blockIdx.x * blockDim.x + threadIdx.x;
  if (i < n) out[i] = (bf16_t)in[i];
}

// in: [R x C] f32 row-major  ->  out: [C x R] bf16 row-major (i.e. transpose)
__global__ void k_cvt_tr_bf16(const float* __restrict__ in, bf16_t* __restrict__ out,
                              int R, int C) {
  int i = blockIdx.x * blockDim.x + threadIdx.x;
  if (i < R * C) {
    int r = i / C, c = i - r * C;
    out[(size_t)c * R + r] = (bf16_t)in[i];
  }
}

// ---------------- bf16 WMMA GEMM, LDS-staged B panel ----------------
// Grid: (M/128) * (N/64) blocks of 256 threads (8 waves).
// Wave w computes rows [blkRow*128 + 16w, +16) x cols [blkCol*64, +64).
__global__ void __launch_bounds__(256)
k_gemm_bf16_wmma(const bf16_t* __restrict__ A,   // [M x K]
                 const bf16_t* __restrict__ Bt,  // [N x K]
                 float*  __restrict__ Cf,
                 bf16_t* __restrict__ Cb,
                 bf16_t* __restrict__ Ct,
                 int M, int N, int K, int apply_elu) {
  __shared__ bf16_t ldsB[64 * LDSB_STRIDE];   // 5 KB

  const int tid    = threadIdx.x;
  const int lane   = tid & 31;
  const int waveId = tid >> 5;               // 0..7
  const int nGrp   = N >> 6;
  const int tm     = (blockIdx.x / nGrp) * 128 + (waveId << 4);
  const int tn     = (blockIdx.x % nGrp) << 6;

  const int half  = lane >> 4;
  const int l16   = lane & 15;
  const int abase = half << 4;     // A K-base: 0 or 16
  const int bbase = half << 4;     // B K-base: 0 or 16

  // cooperative staging assignment: thread -> one 16B chunk of the 64x32 panel
  const int srow = tid >> 2;              // 0..63 (N offset in panel)
  const int sseg = (tid & 3) << 3;        // 0,8,16,24 (K offset)
  const bf16_t* __restrict__ bsrc = Bt + (size_t)(tn + srow) * K + sseg;
  bf16_t* bdst = &ldsB[srow * LDSB_STRIDE + sseg];

  v8f acc0 = {}, acc1 = {}, acc2 = {}, acc3 = {};
  const bf16_t* __restrict__ arow = A + (size_t)(tm + l16) * K;

  for (int k = 0; k < K; k += 32) {
    // stage B panel: global -> regs -> (barrier) -> LDS -> (barrier)
    v8bf bchunk = *(const v8bf*)(bsrc + k);
    __syncthreads();                       // previous iteration's reads done
    *(v8bf*)bdst = bchunk;
    __syncthreads();                       // panel visible to all waves

    // A fragment: two contiguous 16B runs per lane (private rows)
    v8bf alo = *(const v8bf*)(arow + k + abase);
    v8bf ahi = *(const v8bf*)(arow + k + abase + 8);
    v16bf a  = __builtin_shufflevector(alo, ahi,
                 0,1,2,3,4,5,6,7,8,9,10,11,12,13,14,15);
    if (k + 32 < K) __builtin_prefetch(arow + k + 32, 0, 3);  // near prefetch

    // B fragments from LDS (two 16B ds loads each, padded-stride rows)
    v16bf b0, b1, b2, b3;
    {
      v8bf lo, hi;
      lo = *(const v8bf*)&ldsB[( 0 + l16) * LDSB_STRIDE + bbase];
      hi = *(const v8bf*)&ldsB[( 0 + l16) * LDSB_STRIDE + bbase + 8];
      b0 = __builtin_shufflevector(lo, hi, 0,1,2,3,4,5,6,7,8,9,10,11,12,13,14,15);
      lo = *(const v8bf*)&ldsB[(16 + l16) * LDSB_STRIDE + bbase];
      hi = *(const v8bf*)&ldsB[(16 + l16) * LDSB_STRIDE + bbase + 8];
      b1 = __builtin_shufflevector(lo, hi, 0,1,2,3,4,5,6,7,8,9,10,11,12,13,14,15);
      lo = *(const v8bf*)&ldsB[(32 + l16) * LDSB_STRIDE + bbase];
      hi = *(const v8bf*)&ldsB[(32 + l16) * LDSB_STRIDE + bbase + 8];
      b2 = __builtin_shufflevector(lo, hi, 0,1,2,3,4,5,6,7,8,9,10,11,12,13,14,15);
      lo = *(const v8bf*)&ldsB[(48 + l16) * LDSB_STRIDE + bbase];
      hi = *(const v8bf*)&ldsB[(48 + l16) * LDSB_STRIDE + bbase + 8];
      b3 = __builtin_shufflevector(lo, hi, 0,1,2,3,4,5,6,7,8,9,10,11,12,13,14,15);
    }

    acc0 = __builtin_amdgcn_wmma_f32_16x16x32_bf16(false, a, false, b0, (short)0, acc0, false, false);
    acc1 = __builtin_amdgcn_wmma_f32_16x16x32_bf16(false, a, false, b1, (short)0, acc1, false, false);
    acc2 = __builtin_amdgcn_wmma_f32_16x16x32_bf16(false, a, false, b2, (short)0, acc2, false, false);
    acc3 = __builtin_amdgcn_wmma_f32_16x16x32_bf16(false, a, false, b3, (short)0, acc3, false, false);
  }

  #pragma unroll
  for (int t = 0; t < 4; ++t) {
    v8f acc = (t == 0) ? acc0 : (t == 1) ? acc1 : (t == 2) ? acc2 : acc3;
    int gn = tn + (t << 4) + l16;
    #pragma unroll
    for (int r = 0; r < 8; ++r) {
      int gm = tm + r + (half << 3);
      float v = acc[r];
      if (apply_elu) v = v > 0.f ? v : (expf(v) - 1.f);
      if (Cf) Cf[(size_t)gm * N + gn] = v;
      if (Cb) Cb[(size_t)gm * N + gn] = (bf16_t)v;
      if (Ct) Ct[(size_t)gn * M + gm] = (bf16_t)v;
    }
  }
}

// ---------------- per-row attention dots: s_src = h@a_src, s_dst = h@a_dst --

__global__ void k_rowdot(const float* __restrict__ H, const float* __restrict__ asrc,
                         const float* __restrict__ adst, float* __restrict__ ssrc,
                         float* __restrict__ sdst, int Nrows, int D) {
  int wave = (int)((blockIdx.x * blockDim.x + threadIdx.x) >> 5);
  int lane = threadIdx.x & 31;
  if (wave >= Nrows) return;
  const float* h = H + (size_t)wave * D;
  float s0 = 0.f, s1 = 0.f;
  for (int d = lane; d < D; d += 32) {
    float hv = h[d];
    s0 += hv * asrc[d];
    s1 += hv * adst[d];
  }
  #pragma unroll
  for (int off = 16; off; off >>= 1) {
    s0 += __shfl_xor(s0, off, 32);
    s1 += __shfl_xor(s1, off, 32);
  }
  if (lane == 0) { ssrc[wave] = s0; sdst[wave] = s1; }
}

// ---------------- online masked-softmax row stats (max, sum of exp) --------

__global__ void k_softmax_stats(const float* __restrict__ adj,
                                const float* __restrict__ ssrc,
                                const float* __restrict__ sdst,
                                float* __restrict__ rmax, float* __restrict__ rsum,
                                int Nn) {
  int wave = (int)((blockIdx.x * blockDim.x + threadIdx.x) >> 5);
  int lane = threadIdx.x & 31;
  if (wave >= Nn) return;
  float sn = ssrc[wave];
  const float* arow = adj + (size_t)wave * Nn;
  float m = -3.0e38f, s = 0.f;
  for (int mm = lane; mm < Nn; mm += 32) {
    float e = sn + sdst[mm];
    e = e > 0.f ? e : LRELU_SLOPE * e;
    e = (arow[mm] > 0.f) ? e : NEGV;
    float nm = fmaxf(m, e);
    s = s * expf(m - nm) + expf(e - nm);
    m = nm;
  }
  #pragma unroll
  for (int off = 16; off; off >>= 1) {
    float om = __shfl_xor(m, off, 32);
    float os = __shfl_xor(s, off, 32);
    float nm = fmaxf(m, om);
    s = s * expf(m - nm) + os * expf(om - nm);
    m = nm;
  }
  if (lane == 0) { rmax[wave] = m; rsum[wave] = s; }
}

// ---------------- P[n,m] = softmax(e)[n,m] * (beta*amount + (1-beta)*count) -

__global__ void k_attw(const float* __restrict__ adj, const float* __restrict__ amount,
                       const float* __restrict__ count, const float* __restrict__ ssrc,
                       const float* __restrict__ sdst, const float* __restrict__ rmax,
                       const float* __restrict__ rsum, bf16_t* __restrict__ P,
                       int Nn, int log2N) {
  unsigned idx = blockIdx.x * blockDim.x + threadIdx.x;
  if (idx >= (unsigned)Nn * (unsigned)Nn) return;
  int n = idx >> log2N;
  int mcol = idx & (Nn - 1);
  float e = ssrc[n] + sdst[mcol];
  e = e > 0.f ? e : LRELU_SLOPE * e;
  e = (adj[idx] > 0.f) ? e : NEGV;
  float att = expf(e - rmax[n]) / rsum[n];
  float w = BETA * amount[idx] + (1.0f - BETA) * count[idx];
  P[idx] = (bf16_t)(att * w);
}

// ---------------- pair scoring: sigmoid(dot(Y[i], H[j]) + bb) --------------

__global__ void k_pairs(const float* __restrict__ Y, const float* __restrict__ H,
                        const int* __restrict__ pairs, const float* __restrict__ bb,
                        float* __restrict__ out, int nP, int D) {
  int wave = (int)((blockIdx.x * blockDim.x + threadIdx.x) >> 5);
  int lane = threadIdx.x & 31;
  if (wave >= nP) return;
  int i = pairs[2 * wave + 0];
  int j = pairs[2 * wave + 1];
  const float* y = Y + (size_t)i * D;
  const float* h = H + (size_t)j * D;
  float s = 0.f;
  for (int d = lane; d < D; d += 32) s += y[d] * h[d];
  #pragma unroll
  for (int off = 16; off; off >>= 1) s += __shfl_xor(s, off, 32);
  if (lane == 0) out[wave] = 1.f / (1.f + expf(-(s + bb[0])));
}

// ---------------------------------------------------------------------------

extern "C" void kernel_launch(void* const* d_in, const int* in_sizes, int n_in,
                              void* d_out, int out_size, void* d_ws, size_t ws_size,
                              hipStream_t stream) {
  const int N = 4096, IN = 256, HID = 512, OUT = 256;
  const int nP = in_sizes[4] / 2;

  const float* x      = (const float*)d_in[0];
  const float* adj    = (const float*)d_in[1];
  const float* amount = (const float*)d_in[2];
  const float* count  = (const float*)d_in[3];
  const int*   pairs  = (const int*)d_in[4];
  const float* W1     = (const float*)d_in[5];
  const float* a1s    = (const float*)d_in[6];
  const float* a1d    = (const float*)d_in[7];
  const float* W2     = (const float*)d_in[8];
  const float* a2s    = (const float*)d_in[9];
  const float* a2d    = (const float*)d_in[10];
  const float* Wb     = (const float*)d_in[11];
  const float* bb     = (const float*)d_in[12];
  float* out = (float*)d_out;

  // workspace carve-out (256B aligned)
  char* ws = (char*)d_ws;
  size_t off = 0;
  auto alloc = [&](size_t bytes) -> void* {
    void* p = ws + off;
    off = (off + bytes + 255) & ~(size_t)255;
    return p;
  };
  bf16_t* xb    = (bf16_t*)alloc((size_t)N * IN * 2);
  bf16_t* W1t   = (bf16_t*)alloc((size_t)HID * IN * 2);   // [HID x IN]
  bf16_t* W2t   = (bf16_t*)alloc((size_t)OUT * HID * 2);  // [OUT x HID]
  bf16_t* Wbt   = (bf16_t*)alloc((size_t)OUT * OUT * 2);  // [OUT x OUT]
  float*  h1    = (float*) alloc((size_t)N * HID * 4);
  bf16_t* h1t   = (bf16_t*)alloc((size_t)HID * N * 2);    // h1 transposed
  bf16_t* agg1b = (bf16_t*)alloc((size_t)N * HID * 2);
  float*  h2    = (float*) alloc((size_t)N * OUT * 4);
  bf16_t* h2t   = (bf16_t*)alloc((size_t)OUT * N * 2);
  float*  hf    = (float*) alloc((size_t)N * OUT * 4);
  bf16_t* hfb   = (bf16_t*)alloc((size_t)N * OUT * 2);
  float*  Y     = (float*) alloc((size_t)N * OUT * 4);
  float*  s1s   = (float*) alloc((size_t)N * 4);
  float*  s1d   = (float*) alloc((size_t)N * 4);
  float*  rmax  = (float*) alloc((size_t)N * 4);
  float*  rsum  = (float*) alloc((size_t)N * 4);
  bf16_t* Pb    = (bf16_t*)alloc((size_t)N * N * 2);      // 32 MB, reused per layer

  const int T = 256;
  auto cdiv = [](int a, int b) { return (a + b - 1) / b; };
  auto gemmBlocks = [&](int M_, int N_) { return (M_ / 128) * (N_ / 64); };

  // --- weight/input converts ---
  k_cvt_bf16<<<cdiv(N * IN, T), T, 0, stream>>>(x, xb, N * IN);
  k_cvt_tr_bf16<<<cdiv(IN * HID, T), T, 0, stream>>>(W1, W1t, IN, HID);
  k_cvt_tr_bf16<<<cdiv(HID * OUT, T), T, 0, stream>>>(W2, W2t, HID, OUT);
  k_cvt_tr_bf16<<<cdiv(OUT * OUT, T), T, 0, stream>>>(Wb, Wbt, OUT, OUT);

  // --- layer 1 ---
  k_gemm_bf16_wmma<<<gemmBlocks(N, HID), T, 0, stream>>>(
      xb, W1t, h1, nullptr, h1t, N, HID, IN, 0);                 // h1 = x @ W1
  k_rowdot<<<cdiv(N * 32, T), T, 0, stream>>>(h1, a1s, a1d, s1s, s1d, N, HID);
  k_softmax_stats<<<cdiv(N * 32, T), T, 0, stream>>>(adj, s1s, s1d, rmax, rsum, N);
  k_attw<<<cdiv(N * N, T), T, 0, stream>>>(adj, amount, count, s1s, s1d,
                                           rmax, rsum, Pb, N, 12);
  k_gemm_bf16_wmma<<<gemmBlocks(N, HID), T, 0, stream>>>(
      Pb, h1t, nullptr, agg1b, nullptr, N, HID, N, 1);           // elu(P @ h1)

  // --- layer 2 ---
  k_gemm_bf16_wmma<<<gemmBlocks(N, OUT), T, 0, stream>>>(
      agg1b, W2t, h2, nullptr, h2t, N, OUT, HID, 0);             // h2 = agg1 @ W2
  k_rowdot<<<cdiv(N * 32, T), T, 0, stream>>>(h2, a2s, a2d, s1s, s1d, N, OUT);
  k_softmax_stats<<<cdiv(N * 32, T), T, 0, stream>>>(adj, s1s, s1d, rmax, rsum, N);
  k_attw<<<cdiv(N * N, T), T, 0, stream>>>(adj, amount, count, s1s, s1d,
                                           rmax, rsum, Pb, N, 12);
  k_gemm_bf16_wmma<<<gemmBlocks(N, OUT), T, 0, stream>>>(
      Pb, h2t, hf, hfb, nullptr, N, OUT, N, 1);                  // hf = elu(P @ h2)

  // --- pair scoring: Y = hf @ Wb, then gathered dots ---
  k_gemm_bf16_wmma<<<gemmBlocks(N, OUT), T, 0, stream>>>(
      hfb, Wbt, Y, nullptr, nullptr, N, OUT, OUT, 0);
  k_pairs<<<cdiv(nP * 32, T), T, 0, stream>>>(Y, hf, pairs, bb, out, nP, OUT);
}
